// TESTRNN_6837587935697
// MI455X (gfx1250) — matrix operand order repaired and checked by reading, not compiled, not run on hardware
//
#include <hip/hip_runtime.h>
#include <hip/hip_fp16.h>

typedef __attribute__((ext_vector_type(16))) _Float16 v16h;
typedef __attribute__((ext_vector_type(8)))  float    v8f;

#define RNN_B 4096
#define RNN_T 2048
#define RNN_H 20
#define LDS_STRIDE 36   // row stride in floats; spreads LDS banks, holds j<32

__device__ __forceinline__ float fast_tanh(float x) {
#if __has_builtin(__builtin_amdgcn_tanhf)
    return __builtin_amdgcn_tanhf(x);           // v_tanh_f32 (gfx1250 trans op)
#else
    // tanh(x) = 1 - 2/(e^{2x}+1)
    float e = __expf(2.0f * x);
    return 1.0f - 2.0f * __builtin_amdgcn_rcpf(e + 1.0f);
#endif
}

// One wave (32 threads) per block; each block owns one 16-row batch tile and
// runs BOTH directions' recurrences interleaved for ILP.
__global__ void __launch_bounds__(32)
rnn_scan_kernel(const float* __restrict__ x,
                const float* __restrict__ Wih_f, const float* __restrict__ Whh_f,
                const float* __restrict__ bih_f, const float* __restrict__ bhh_f,
                const float* __restrict__ Wih_b, const float* __restrict__ Whh_b,
                const float* __restrict__ bih_b, const float* __restrict__ bhh_b,
                float* __restrict__ hbuf /* [2][B][H] */)
{
    __shared__ float sh[2][16 * LDS_STRIDE];

    const int lane = threadIdx.x;   // 0..31
    const int n    = lane & 15;     // lane-in-half: N column (WMMA) / batch row (row domain)
    const int hi   = lane >> 4;     // half-wave
    const int b0   = blockIdx.x * 16;

    const float* Wih[2] = {Wih_f, Wih_b};
    const float* Whh[2] = {Whh_f, Whh_b};
    const float* bih[2] = {bih_f, bih_b};
    const float* bhh[2] = {bhh_f, bhh_b};

    // ---- Build constant B operands (W_hh^T), per direction, two N tiles. ----
    // B 32x16 f16 layout: half i of lane l holds B(k = i<8 ? i + 8*(l>>4) : 16 + 8*(l>>4) + (i-8), n = l&15).
    // B(k, j) = W_hh[j][k]  (since s = h @ W_hh^T).
    v16h Bt0[2], Bt1[2];
    #pragma unroll
    for (int d = 0; d < 2; ++d) {
        #pragma unroll
        for (int i = 0; i < 16; ++i) {
            int k  = (i < 8) ? (i + 8 * hi) : (16 + 8 * hi + (i - 8));
            int j1 = 16 + n;
            float v0 = (k < RNN_H)                 ? Whh[d][n  * RNN_H + k] : 0.0f;
            float v1 = (k < RNN_H && j1 < RNN_H)   ? Whh[d][j1 * RNN_H + k] : 0.0f;
            Bt0[d][i] = (_Float16)v0;
            Bt1[d][i] = (_Float16)v1;
        }
    }

    // ---- Per-lane slot constants (A-matrix 16x32 f16 layout). ----
    // A half i of lane l holds A(m = l&15, K): i<8 -> K = i + 8*hi;
    // i in 8..11 -> K = 16 + (i-8) for hi==0 (hi==1 halves are K>=24: padding).
    int   slotK[12];
    bool  svalid[12];
    float wk[2][12], ck[2][12];
    #pragma unroll
    for (int s = 0; s < 12; ++s) {
        int K; bool v;
        if (s < 8) { K = s + 8 * hi;             v = true; }
        else       { K = (hi == 0) ? (8 + s) : 0; v = (hi == 0); }
        slotK[s]  = K;
        svalid[s] = v;
        #pragma unroll
        for (int d = 0; d < 2; ++d) {
            wk[d][s] = v ? Wih[d][K] : 0.0f;
            ck[d][s] = v ? (bih[d][K] + bhh[d][K]) : 0.0f;
        }
    }

    const float* xrow = x + (size_t)(b0 + n) * RNN_T;  // per-lane sequence row

    v16h a[2];                    // h state as f16 A operand (zero-padded K>=20)
    #pragma unroll
    for (int d = 0; d < 2; ++d)
        #pragma unroll
        for (int i = 0; i < 16; ++i) a[d][i] = (_Float16)0.0f;

    float actv[2][12];
    #pragma unroll
    for (int d = 0; d < 2; ++d)
        #pragma unroll
        for (int s = 0; s < 12; ++s) actv[d][s] = 0.0f;

    v8f cz = {};  // zero accumulator (loop-invariant)

    for (int t = 0; t < RNN_T; ++t) {
        float xv[2];
        xv[0] = xrow[t];                 // forward
        xv[1] = xrow[RNN_T - 1 - t];     // backward (x[:, ::-1])

        // ---- s = h @ W_hh^T via WMMA; scatter D into LDS row-major [m][j]. ----
        #pragma unroll
        for (int d = 0; d < 2; ++d) {
            v8f d0 = __builtin_amdgcn_wmma_f32_16x16x32_f16(
                         false, a[d], false, Bt0[d], (short)0, cz, false, false);
            v8f d1 = __builtin_amdgcn_wmma_f32_16x16x32_f16(
                         false, a[d], false, Bt1[d], (short)0, cz, false, false);
            #pragma unroll
            for (int r = 0; r < 8; ++r) {
                int m = r + 8 * hi;                          // D: vgpr r, half hi -> row m
                sh[d][m * LDS_STRIDE + n]      = d0[r];      // j = n      (0..15)
                sh[d][m * LDS_STRIDE + 16 + n] = d1[r];      // j = 16 + n (16..19 valid)
            }
        }
        __syncthreads();  // single-wave WG: lowers to a cheap dscnt wait

        // ---- Row domain: lane = batch row n. xp add + tanh + repack to A. ----
        #pragma unroll
        for (int d = 0; d < 2; ++d) {
            #pragma unroll
            for (int s = 0; s < 12; ++s) {
                float sv  = sh[d][n * LDS_STRIDE + slotK[s]];
                float pre = fmaf(xv[d], wk[d][s], ck[d][s]) + sv;
                float av  = fast_tanh(pre);
                av        = svalid[s] ? av : 0.0f;
                actv[d][s] = av;
                a[d][s]    = (_Float16)av;   // halves 12..15 stay zero forever
            }
        }
        __syncthreads();
    }

    // ---- Write final hidden states h_T to workspace: hbuf[d][b][K]. ----
    #pragma unroll
    for (int d = 0; d < 2; ++d) {
        #pragma unroll
        for (int s = 0; s < 12; ++s) {
            if (svalid[s])
                hbuf[(size_t)d * RNN_B * RNN_H + (size_t)(b0 + n) * RNN_H + slotK[s]]
                    = actv[d][s];
        }
    }
}

// hn = stack([hf, hb], 0).reshape(-1, 40): row k<2048 pairs hf rows (2k, 2k+1);
// row k>=2048 pairs hb rows (2(k-2048), 2(k-2048)+1). out = hn @ W_fc^T + b_fc.
__global__ void __launch_bounds__(256)
rnn_fc_kernel(const float* __restrict__ hbuf, const float* __restrict__ Wfc,
              const float* __restrict__ bfc, float* __restrict__ out)
{
    int k = blockIdx.x * blockDim.x + threadIdx.x;
    if (k >= RNN_B) return;
    int d  = (k >= RNN_B / 2) ? 1 : 0;
    int kk = k - d * (RNN_B / 2);
    const float* h0 = hbuf + (size_t)d * RNN_B * RNN_H + (size_t)(2 * kk) * RNN_H;
    float acc = bfc[0];
    #pragma unroll
    for (int i = 0; i < RNN_H; ++i) acc = fmaf(h0[i],          Wfc[i],          acc);
    #pragma unroll
    for (int i = 0; i < RNN_H; ++i) acc = fmaf(h0[RNN_H + i],  Wfc[RNN_H + i],  acc);
    out[k] = acc;
}

extern "C" void kernel_launch(void* const* d_in, const int* in_sizes, int n_in,
                              void* d_out, int out_size, void* d_ws, size_t ws_size,
                              hipStream_t stream) {
    const float* x     = (const float*)d_in[0];
    const float* Wih_f = (const float*)d_in[1];
    const float* Whh_f = (const float*)d_in[2];
    const float* bih_f = (const float*)d_in[3];
    const float* bhh_f = (const float*)d_in[4];
    const float* Wih_b = (const float*)d_in[5];
    const float* Whh_b = (const float*)d_in[6];
    const float* bih_b = (const float*)d_in[7];
    const float* bhh_b = (const float*)d_in[8];
    const float* Wfc   = (const float*)d_in[9];
    const float* bfc   = (const float*)d_in[10];

    float* hbuf = (float*)d_ws;            // 2 * 4096 * 20 floats = 640 KiB
    float* out  = (float*)d_out;

    rnn_scan_kernel<<<RNN_B / 16, 32, 0, stream>>>(
        x, Wih_f, Whh_f, bih_f, bhh_f, Wih_b, Whh_b, bih_b, bhh_b, hbuf);
    rnn_fc_kernel<<<(RNN_B + 255) / 256, 256, 0, stream>>>(hbuf, Wfc, bfc, out);
}
